// Block_3453153706485
// MI455X (gfx1250) — matrix operand order repaired and checked by reading, not compiled, not run on hardware
//
#include <hip/hip_runtime.h>
#include <hip/hip_bf16.h>
#include <math.h>

// ---------------------------------------------------------------------------
// Transformer block for MI455X (gfx1250), bf16 WMMA everywhere.
//   B=4 S=1024 D=1024 H=16 HD=64 FF=4096
// ---------------------------------------------------------------------------

#define B_  4
#define S_  1024
#define D_  1024
#define H_  16
#define HD_ 64
#define FF_ 4096
#define M_  (B_ * S_)   // 4096 token rows

typedef __attribute__((ext_vector_type(16))) __bf16 v16bf;
typedef __attribute__((ext_vector_type(8)))  __bf16 v8bf;
typedef __attribute__((ext_vector_type(8)))  float  v8f;

static __device__ inline v16bf join16(v8bf lo, v8bf hi) {
    v16bf r;
#pragma unroll
    for (int i = 0; i < 8; ++i) { r[i] = lo[i]; r[8 + i] = hi[i]; }
    return r;
}

// D = A(16x32 bf16) * B(32x16 bf16) + C(16x16 f32)
static __device__ inline v8f wmma_bf16(v16bf a, v16bf b, v8f c) {
    return __builtin_amdgcn_wmma_f32_16x16x32_bf16(
        /*neg_a=*/false, a, /*neg_b=*/false, b,
        /*c_mod=*/(short)0, c, /*reuse_a=*/false, /*reuse_b=*/false);
}

// ---------------------------------------------------------------------------
// fp32 -> bf16 conversion (weights / simple tensors)
// ---------------------------------------------------------------------------
__global__ void cvt_bf16_kernel(const float* __restrict__ in,
                                __bf16* __restrict__ out, int n) {
    int i = blockIdx.x * 256 + threadIdx.x;
    if (i < n) out[i] = (__bf16)in[i];
}

// ---------------------------------------------------------------------------
// LayerNorm over D=1024, one 256-thread block per row, bf16 output.
// ---------------------------------------------------------------------------
__global__ void layernorm_bf16_kernel(const float* __restrict__ x,
                                      const float* __restrict__ w,
                                      const float* __restrict__ b,
                                      __bf16* __restrict__ out) {
    __shared__ float red[256];
    const int row = blockIdx.x;
    const int t   = threadIdx.x;
    const float* xr = x + (size_t)row * D_;

    float vals[4];
    float s = 0.f;
#pragma unroll
    for (int i = 0; i < 4; ++i) { vals[i] = xr[t + 256 * i]; s += vals[i]; }
    red[t] = s; __syncthreads();
    for (int o2 = 128; o2 > 0; o2 >>= 1) {
        if (t < o2) red[t] += red[t + o2];
        __syncthreads();
    }
    const float mean = red[0] * (1.0f / D_);
    __syncthreads();

    float vs = 0.f;
#pragma unroll
    for (int i = 0; i < 4; ++i) { float d = vals[i] - mean; vs += d * d; }
    red[t] = vs; __syncthreads();
    for (int o2 = 128; o2 > 0; o2 >>= 1) {
        if (t < o2) red[t] += red[t + o2];
        __syncthreads();
    }
    const float var = red[0] * (1.0f / D_);
    const float rs  = rsqrtf(var + 1e-5f);

#pragma unroll
    for (int i = 0; i < 4; ++i) {
        int c = t + 256 * i;
        out[(size_t)row * D_ + c] = (__bf16)((vals[i] - mean) * rs * w[c] + b[c]);
    }
}

// ---------------------------------------------------------------------------
// RoPE on (B,S,H,HD) fp32 -> bf16 (interleaved even/odd pairs).
// ---------------------------------------------------------------------------
__global__ void rope_bf16_kernel(const float* __restrict__ in,
                                 const float* __restrict__ cosb,
                                 const float* __restrict__ sinb,
                                 __bf16* __restrict__ out) {
    int i  = blockIdx.x * 256 + threadIdx.x;    // pair index
    int d2 = i & 31;                            // HD/2 = 32
    int s  = (i >> 9) & 1023;
    float c  = cosb[s * 32 + d2];
    float sn = sinb[s * 32 + d2];
    float e = in[2 * i], o = in[2 * i + 1];
    out[2 * i]     = (__bf16)(e * c  - o * sn);
    out[2 * i + 1] = (__bf16)(e * sn + o * c);
}

// ---------------------------------------------------------------------------
// Tiled WMMA GEMM: C[M x N] = A[M x K](bf16,row-major) * W[K x N](bf16,row-major)
// Block tile 128x128, 256 threads = 8 waves (4 row-waves x 2 col-waves),
// each wave owns a 32x64 strip: 2 A-frags x 4 B-frags = 8 WMMAs per K-step.
// A staged row-major in LDS; W staged K-transposed so B-fragments are
// contiguous ds_load_b128s. Next K-tile prefetched via global_prefetch_b8.
// Epilogues: 0 = store f32; 1 = store bf16; 2 = GELU(exact)->bf16;
//            3 = res + C -> f32
// ---------------------------------------------------------------------------
__global__ void gemm_bf16_kernel(const __bf16* __restrict__ A,
                                 const __bf16* __restrict__ W,
                                 int K, int N,
                                 float* __restrict__ outf,
                                 __bf16* __restrict__ outb,
                                 const float* __restrict__ res,
                                 int epi) {
    __shared__ __align__(32) __bf16 sA [128 * 32];   // [m][k]   8 KB
    __shared__ __align__(32) __bf16 sWt[128 * 32];   // [n][k]   8 KB

    const int t    = threadIdx.x;
    const int lane = t & 31;
    const int wave = t >> 5;
    const int g    = lane >> 4;        // half-wave group (WMMA lane group)
    const int ln   = lane & 15;
    const int wm   = wave & 3;         // wave row (32 rows each)
    const int wn   = wave >> 2;        // wave col (64 cols each)
    const int mBase = blockIdx.y * 128;
    const int nBase = blockIdx.x * 128;

    // staging coordinates
    const int am = t >> 1,  ak0 = (t & 1) << 4;   // A: 32B per thread
    const int wk = t >> 3,  wn0 = (t & 7) << 4;   // W: 32B per thread

    v8f acc[2][4];
#pragma unroll
    for (int mi = 0; mi < 2; ++mi)
#pragma unroll
        for (int f = 0; f < 4; ++f)
#pragma unroll
            for (int v = 0; v < 8; ++v) acc[mi][f][v] = 0.f;

    for (int kb = 0; kb < K; kb += 32) {
        // stage A tile: 128x32, 16 elems (32B) per thread
        *(v16bf*)&sA[am * 32 + ak0] =
            *(const v16bf*)&A[(size_t)(mBase + am) * K + kb + ak0];
        // stage W tile transposed: 32(K) x 128(N) -> sWt[n][k]
        {
            v16bf wrow = *(const v16bf*)&W[(size_t)(kb + wk) * N + nBase + wn0];
#pragma unroll
            for (int i = 0; i < 16; ++i) sWt[(wn0 + i) * 32 + wk] = wrow[i];
        }
        // prefetch next K tile (emits global_prefetch_b8)
        if (kb + 32 < K) {
            __builtin_prefetch(&A[(size_t)(mBase + am) * K + kb + 32 + ak0], 0, 1);
            __builtin_prefetch(&W[(size_t)(kb + 32 + wk) * N + nBase + wn0], 0, 1);
        }
        __syncthreads();

        // A fragments: lane ln = row M; elems K = g*16 + e (contiguous)
        v16bf af[2];
#pragma unroll
        for (int mi = 0; mi < 2; ++mi) {
            int mrow = wm * 32 + mi * 16 + ln;
            af[mi] = *(const v16bf*)&sA[mrow * 32 + g * 16];
        }
#pragma unroll
        for (int nf = 0; nf < 4; ++nf) {
            // B fragment: lane ln = col N; elems K = g*16 + e (contiguous)
            int n = wn * 64 + nf * 16 + ln;
            v16bf bf = *(const v16bf*)&sWt[n * 32 + g * 16];
#pragma unroll
            for (int mi = 0; mi < 2; ++mi)
                acc[mi][nf] = wmma_bf16(af[mi], bf, acc[mi][nf]);
        }
        __syncthreads();
    }

    // C layout: lane = (M/8)*16 + N%16, elem v = M%8
#pragma unroll
    for (int mi = 0; mi < 2; ++mi) {
#pragma unroll
        for (int nf = 0; nf < 4; ++nf) {
#pragma unroll
            for (int v = 0; v < 8; ++v) {
                int row = mBase + wm * 32 + mi * 16 + v + 8 * g;
                int col = nBase + wn * 64 + nf * 16 + ln;
                size_t idx = (size_t)row * N + col;
                float xv = acc[mi][nf][v];
                if (epi == 0)      outf[idx] = xv;
                else if (epi == 1) outb[idx] = (__bf16)xv;
                else if (epi == 2) outb[idx] =
                    (__bf16)(0.5f * xv * (1.0f + erff(xv * 0.70710678f)));
                else               outf[idx] = res[idx] + xv;
            }
        }
    }
}

// ---------------------------------------------------------------------------
// Flash attention: block = 4 waves sharing one (b,h); each wave owns a
// 16-row q tile. K/V tiles (32 keys) are cooperatively staged in LDS
// (V stored K-transposed) so every WMMA operand is a contiguous ds_load_b128.
// Online softmax uses cross-lane shfl reductions inside 16-lane halves;
// P converts C-layout -> A-layout via a per-wave LDS bounce (in-wave DS
// ordering, no barrier needed). Output bf16 in (b,s,h,hd).
// ---------------------------------------------------------------------------
__global__ void attn_kernel(const __bf16* __restrict__ q,
                            const __bf16* __restrict__ k,
                            const __bf16* __restrict__ v,
                            const int* __restrict__ mask,
                            __bf16* __restrict__ out) {
    __shared__ __align__(32) __bf16 sK [32 * 64];    // [kpos][hd]   4 KB
    __shared__ __align__(32) __bf16 sVt[64 * 32];    // [hd][kpos]   4 KB
    __shared__ __align__(32) __bf16 sP [4][16 * 32]; // per-wave P   4 KB

    const int t     = threadIdx.x;
    const int lane  = t & 31;
    const int wslot = t >> 5;
    const int g     = lane >> 4;
    const int ln    = lane & 15;
    const int wid   = blockIdx.x * 4 + wslot;
    const int qt = wid & 63;            // S/16 = 64 q tiles; block-uniform b,h
    const int h  = (wid >> 6) & 15;
    const int b  = wid >> 10;
    const int qbase = qt * 16;

    // staging coordinates: 128 threads cover 32 rows x 64 cols, 32B each
    const int sr = t >> 2, sc0 = (t & 3) << 4;

    // Q A-fragments (K = HD = 64 -> 2 chunks of 32)
    const __bf16* qp = q + ((size_t)(b * S_ + qbase + ln)) * D_ + h * HD_;
    v16bf aq[2];
#pragma unroll
    for (int c = 0; c < 2; ++c)
        aq[c] = *(const v16bf*)&qp[c * 32 + g * 16];

    float mo[8], ls[8];
    v8f o[4];
#pragma unroll
    for (int vv = 0; vv < 8; ++vv) { mo[vv] = -3e38f; ls[vv] = 0.f; }
#pragma unroll
    for (int f = 0; f < 4; ++f)
#pragma unroll
        for (int vv = 0; vv < 8; ++vv) o[f][vv] = 0.f;

    const int* mrow = mask + b * S_;

    for (int kb0 = 0; kb0 < S_; kb0 += 32) {
        __syncthreads();   // previous step's LDS reads complete
        // stage K tile row-major (B-frag for scores reads hd contiguously)
        *(v16bf*)&sK[sr * 64 + sc0] =
            *(const v16bf*)&k[((size_t)(b * S_ + kb0 + sr)) * D_ + h * HD_ + sc0];
        // stage V tile transposed (B-frag for P*V reads kpos contiguously)
        {
            v16bf vrow =
                *(const v16bf*)&v[((size_t)(b * S_ + kb0 + sr)) * D_ + h * HD_ + sc0];
#pragma unroll
            for (int i = 0; i < 16; ++i) sVt[(sc0 + i) * 32 + sr] = vrow[i];
        }
        __syncthreads();

        // scores: S[16q x 32k] = Q(16x64) . K^T
        v8f scr[2];
#pragma unroll
        for (int j = 0; j < 2; ++j)
#pragma unroll
            for (int vv = 0; vv < 8; ++vv) scr[j][vv] = 0.f;

#pragma unroll
        for (int j = 0; j < 2; ++j) {
            int kp = j * 16 + ln;           // this lane's key column (local)
#pragma unroll
            for (int c = 0; c < 2; ++c) {
                v16bf bk = *(const v16bf*)&sK[kp * 64 + c * 32 + g * 16];
                scr[j] = wmma_bf16(aq[c], bk, scr[j]);
            }
            float bias = mrow[kb0 + kp] ? -1e30f : 0.0f;   // -inf surrogate
#pragma unroll
            for (int vv = 0; vv < 8; ++vv)
                scr[j][vv] = scr[j][vv] * 0.125f + bias;   // 1/sqrt(64)
        }

        // online softmax (rows live in 16-lane halves; elem vv = row M%8)
#pragma unroll
        for (int vv = 0; vv < 8; ++vv) {
            float rm = fmaxf(scr[0][vv], scr[1][vv]);
#pragma unroll
            for (int mk = 1; mk < 16; mk <<= 1)
                rm = fmaxf(rm, __shfl_xor(rm, mk, 32));
            float mn = fmaxf(mo[vv], rm);
            float alpha = __expf(mo[vv] - mn);
            float p0 = __expf(scr[0][vv] - mn);
            float p1 = __expf(scr[1][vv] - mn);
            scr[0][vv] = p0; scr[1][vv] = p1;
            float rs = p0 + p1;
#pragma unroll
            for (int mk = 1; mk < 16; mk <<= 1)
                rs += __shfl_xor(rs, mk, 32);
            ls[vv] = alpha * ls[vv] + rs;
            mo[vv] = mn;
#pragma unroll
            for (int f = 0; f < 4; ++f) o[f][vv] *= alpha;
        }

        // P (16x32) C-layout -> per-wave LDS row-major -> A fragment.
        // Same-wave DS ops complete in order; no barrier needed.
#pragma unroll
        for (int j = 0; j < 2; ++j)
#pragma unroll
            for (int vv = 0; vv < 8; ++vv)
                sP[wslot][(vv + 8 * g) * 32 + j * 16 + ln] = (__bf16)scr[j][vv];
        v16bf ap = *(const v16bf*)&sP[wslot][ln * 32 + g * 16];

        // O += P(16x32) . V(32x64)
#pragma unroll
        for (int f = 0; f < 4; ++f) {
            v16bf bv = *(const v16bf*)&sVt[(f * 16 + ln) * 32 + g * 16];
            o[f] = wmma_bf16(ap, bv, o[f]);
        }
    }

    // normalize and store bf16 (b,s,h,hd)
#pragma unroll
    for (int vv = 0; vv < 8; ++vv) {
        float inv = 1.0f / ls[vv];
        int row = b * S_ + qbase + vv + 8 * g;
#pragma unroll
        for (int f = 0; f < 4; ++f)
            out[(size_t)row * D_ + h * HD_ + f * 16 + ln] =
                (__bf16)(o[f][vv] * inv);
    }
}

// ---------------------------------------------------------------------------
// Host-side orchestration
// ---------------------------------------------------------------------------
extern "C" void kernel_launch(void* const* d_in, const int* in_sizes, int n_in,
                              void* d_out, int out_size, void* d_ws, size_t ws_size,
                              hipStream_t stream) {
    (void)in_sizes; (void)n_in; (void)out_size; (void)ws_size;

    const float* x    = (const float*)d_in[0];
    const int*   mask = (const int*)  d_in[1];
    const float* fcos = (const float*)d_in[2];
    const float* fsin = (const float*)d_in[3];
    const float* wq   = (const float*)d_in[4];
    const float* wk   = (const float*)d_in[5];
    const float* wv   = (const float*)d_in[6];
    const float* wo   = (const float*)d_in[7];
    const float* w1   = (const float*)d_in[8];
    const float* w2   = (const float*)d_in[9];
    const float* ln1w = (const float*)d_in[10];
    const float* ln1b = (const float*)d_in[11];
    const float* ln2w = (const float*)d_in[12];
    const float* ln2b = (const float*)d_in[13];
    float* out = (float*)d_out;

    // workspace partition (peak ~112 MB with reuse)
    char* ws = (char*)d_ws;
    const size_t MB = 1ull << 20;
    __bf16* wqb = (__bf16*)(ws +  0 * MB);   // 2 MB
    __bf16* wkb = (__bf16*)(ws +  2 * MB);   // 2 MB
    __bf16* wvb = (__bf16*)(ws +  4 * MB);   // 2 MB
    __bf16* wob = (__bf16*)(ws +  6 * MB);   // 2 MB
    __bf16* w1b = (__bf16*)(ws +  8 * MB);   // 8 MB
    __bf16* w2b = (__bf16*)(ws + 16 * MB);   // 8 MB
    __bf16* hb  = (__bf16*)(ws + 24 * MB);   // 8 MB (ln1 out; reused as ln2 out)
    float*  qf  = (float*) (ws + 32 * MB);   // 16 MB (dead after RoPE)
    float*  kf  = (float*) (ws + 48 * MB);   // 16 MB (dead after RoPE)
    __bf16* vb  = (__bf16*)(ws + 64 * MB);   // 8 MB
    __bf16* qb  = (__bf16*)(ws + 72 * MB);   // 8 MB
    __bf16* kb  = (__bf16*)(ws + 80 * MB);   // 8 MB
    __bf16* ab  = (__bf16*)(ws + 88 * MB);   // 8 MB (attention out)
    float*  x1  = (float*) (ws + 96 * MB);   // 16 MB (x + attn, live to end)
    __bf16* h2b = hb;                        // reuse
    __bf16* gb  = (__bf16*)(ws + 32 * MB);   // 32 MB, reuses qf+kf region

    // 1) weights -> bf16
    cvt_bf16_kernel<<<(D_ * D_) / 256, 256, 0, stream>>>(wq, wqb, D_ * D_);
    cvt_bf16_kernel<<<(D_ * D_) / 256, 256, 0, stream>>>(wk, wkb, D_ * D_);
    cvt_bf16_kernel<<<(D_ * D_) / 256, 256, 0, stream>>>(wv, wvb, D_ * D_);
    cvt_bf16_kernel<<<(D_ * D_) / 256, 256, 0, stream>>>(wo, wob, D_ * D_);
    cvt_bf16_kernel<<<(D_ * FF_) / 256, 256, 0, stream>>>(w1, w1b, D_ * FF_);
    cvt_bf16_kernel<<<(FF_ * D_) / 256, 256, 0, stream>>>(w2, w2b, FF_ * D_);

    // 2) h = LN1(x)
    layernorm_bf16_kernel<<<M_, 256, 0, stream>>>(x, ln1w, ln1b, hb);

    // 3) q,k (f32 for RoPE), v (bf16 directly)
    dim3 gD(D_ / 128, M_ / 128);
    gemm_bf16_kernel<<<gD, 256, 0, stream>>>(hb, wqb, D_, D_, qf, nullptr, nullptr, 0);
    gemm_bf16_kernel<<<gD, 256, 0, stream>>>(hb, wkb, D_, D_, kf, nullptr, nullptr, 0);
    gemm_bf16_kernel<<<gD, 256, 0, stream>>>(hb, wvb, D_, D_, nullptr, vb, nullptr, 1);

    // 4) RoPE -> bf16 q,k
    const int npair = B_ * S_ * H_ * (HD_ / 2);
    rope_bf16_kernel<<<npair / 256, 256, 0, stream>>>(qf, fcos, fsin, qb);
    rope_bf16_kernel<<<npair / 256, 256, 0, stream>>>(kf, fcos, fsin, kb);

    // 5) flash attention -> bf16
    attn_kernel<<<(B_ * H_ * (S_ / 16)) / 4, 128, 0, stream>>>(qb, kb, vb, mask, ab);

    // 6) x1 = x + attn @ wo
    gemm_bf16_kernel<<<gD, 256, 0, stream>>>(ab, wob, D_, D_, x1, nullptr, x, 3);

    // 7) h2 = LN2(x1)
    layernorm_bf16_kernel<<<M_, 256, 0, stream>>>(x1, ln2w, ln2b, h2b);

    // 8) g = GELU(h2 @ w1) -> bf16
    dim3 gF(FF_ / 128, M_ / 128);
    gemm_bf16_kernel<<<gF, 256, 0, stream>>>(h2b, w1b, D_, FF_, nullptr, gb, nullptr, 2);

    // 9) out = x1 + g @ w2
    gemm_bf16_kernel<<<gD, 256, 0, stream>>>(gb, w2b, FF_, D_, out, nullptr, x1, 3);
}
